// HyperModel_65755949301857
// MI455X (gfx1250) — matrix operand order repaired and checked by reading, not compiled, not run on hardware
//
#include <hip/hip_runtime.h>

typedef __attribute__((ext_vector_type(16))) _Float16 v16h;
typedef __attribute__((ext_vector_type(8)))  _Float16 v8h;
typedef __attribute__((ext_vector_type(8)))  float    v8f;

#define U_CNT 339
#define S_CNT 5825
#define N_CNT 6164
#define D_DIM 128
#define H_DIM 64
#define R_CNT 32
#define B_CNT 500000
#define E_CNT 400000
#define LN_EPS 1e-5f

// ---------------------------------------------------------------- utilities
__global__ void concat_kernel(const float* __restrict__ uE,
                              const float* __restrict__ iE,
                              float* __restrict__ emb) {
  int i = blockIdx.x * 256 + threadIdx.x;
  if (i < N_CNT * D_DIM)
    emb[i] = (i < U_CNT * D_DIM) ? uE[i] : iE[i - U_CNT * D_DIM];
}

__global__ void zero_kernel(float* __restrict__ p, int n) {
  int i = blockIdx.x * 256 + threadIdx.x;
  if (i < n) p[i] = 0.0f;
}

// -------------------------------------------------- COO SpMM (scatter-atomic)
// lanes 0..31 of a wave share one edge; each lane handles 4 of 128 dims.
__global__ void spmm_scatter(const float* __restrict__ in,
                             float* __restrict__ out,
                             const float* __restrict__ vals,
                             const int* __restrict__ rows,
                             const int* __restrict__ cols) {
  int gid = blockIdx.x * 256 + threadIdx.x;
  int e = gid >> 5;
  if (e >= E_CNT) return;
  int part = gid & 31;
  int c = cols[e];
  int r = rows[e];
  float v = vals[e];
  float4 x = ((const float4*)(in + (size_t)c * D_DIM))[part];
  float* o = out + (size_t)r * D_DIM + part * 4;
  atomicAdd(o + 0, v * x.x);
  atomicAdd(o + 1, v * x.y);
  atomicAdd(o + 2, v * x.z);
  atomicAdd(o + 3, v * x.w);
}

// ------------------------------------------- fold hyper + W1: T = H @ W1part
__global__ void tmat_kernel(const float* __restrict__ uH,
                            const float* __restrict__ iH,
                            const float* __restrict__ W1,
                            float* __restrict__ Tu, float* __restrict__ Ti) {
  int idx = blockIdx.x * 256 + threadIdx.x;
  if (idx >= 2 * R_CNT * H_DIM) return;
  int which = idx / (R_CNT * H_DIM);
  int rem = idx % (R_CNT * H_DIM);
  int r = rem / H_DIM, j = rem % H_DIM;
  const float* Hp = which ? iH : uH;
  const float* Wp = W1 + (which ? (size_t)D_DIM * H_DIM : 0);
  float acc = 0.0f;
  for (int k = 0; k < D_DIM; ++k)
    acc += Hp[r * D_DIM + k] * Wp[k * H_DIM + j];
  (which ? Ti : Tu)[rem] = acc;
}

// M = H^T @ T   [128 x 64]
__global__ void mmat_kernel(const float* __restrict__ uH,
                            const float* __restrict__ iH,
                            const float* __restrict__ Tu,
                            const float* __restrict__ Ti,
                            float* __restrict__ Mu, float* __restrict__ Mi) {
  int idx = blockIdx.x * 256 + threadIdx.x;
  if (idx >= 2 * D_DIM * H_DIM) return;
  int which = idx / (D_DIM * H_DIM);
  int rem = idx % (D_DIM * H_DIM);
  int i = rem / H_DIM, j = rem % H_DIM;
  const float* Hp = which ? iH : uH;
  const float* Tp = which ? Ti : Tu;
  float acc = 0.0f;
  for (int r = 0; r < R_CNT; ++r)
    acc += Hp[r * D_DIM + i] * Tp[r * H_DIM + j];
  (which ? Mi : Mu)[rem] = acc;
}

// P = e @ M  (per side)
__global__ void proj_kernel(const float* __restrict__ emb,
                            const float* __restrict__ Mu,
                            const float* __restrict__ Mi,
                            float* __restrict__ Pu, float* __restrict__ Pi) {
  int idx = blockIdx.x * 256 + threadIdx.x;
  if (idx >= N_CNT * H_DIM) return;
  int n = idx / H_DIM, j = idx % H_DIM;
  const float* M;
  float* P;
  int nn;
  if (n < U_CNT) { M = Mu; P = Pu; nn = n; }
  else           { M = Mi; P = Pi; nn = n - U_CNT; }
  const float* er = emb + (size_t)n * D_DIM;
  float acc = 0.0f;
#pragma unroll 8
  for (int k = 0; k < D_DIM; ++k) acc += er[k] * M[k * H_DIM + j];
  P[(size_t)nn * H_DIM + j] = acc;
}

// ------------------------- pre-swizzle W2 (f32 64x64) into WMMA B fragments
// fragment layout: element e of lane L holds B(K,N) with
//   K = q*32 + e + (e>=8 ? 8 : 0) + (L>=16 ? 8 : 0),  N = t*16 + (L&15)   (per ISA 7.12.2)
// stored as W2s[((t*2+q)*32 + lane)*16 + e]
__global__ void w2_swizzle(const float* __restrict__ W2,
                           _Float16* __restrict__ W2s) {
  int idx = blockIdx.x * 256 + threadIdx.x;
  if (idx >= 4096) return;
  int e = idx & 15;
  int lane = (idx >> 4) & 31;
  int q = (idx >> 9) & 1;
  int t = idx >> 10;
  int K = q * 32 + e + ((e >= 8) ? 8 : 0) + ((lane >= 16) ? 8 : 0);
  int Nn = t * 16 + (lane & 15);
  W2s[idx] = (_Float16)W2[K * H_DIM + Nn];
}

// ------------------------------------------------------------- fused MLP
// 128 threads (4 waves) per block, 128 batch rows per block.
__launch_bounds__(128)
__global__ void mlp_kernel(const float* __restrict__ Pu,
                           const float* __restrict__ Pi,
                           const _Float16* __restrict__ W2s,
                           const int* __restrict__ uIdx,
                           const int* __restrict__ sIdx,
                           const float* __restrict__ b1,
                           const float* __restrict__ g1,
                           const float* __restrict__ be1,
                           const float* __restrict__ b2,
                           const float* __restrict__ g2,
                           const float* __restrict__ be2,
                           const float* __restrict__ W3,
                           const float* __restrict__ b3,
                           float* __restrict__ out) {
  __shared__ __align__(16) _Float16 hbuf[128 * 72];  // 72-half stride: bank-safe
  const int tid = threadIdx.x;
  const int lane = tid & 31;
  const int wave = tid >> 5;
  const int row = blockIdx.x * 128 + tid;

  // ---- phase 1: h1 = relu(LN(P_u[u] + P_i[s] + b1)) per thread ----
  float v[64];
  if (row < B_CNT) {
    int u = uIdx[row];
    int s = sIdx[row];
    const float4* pu = (const float4*)(Pu + (size_t)u * H_DIM);
    const float4* pi = (const float4*)(Pi + (size_t)s * H_DIM);
    const float4* pb = (const float4*)b1;
#pragma unroll
    for (int j = 0; j < 16; ++j) {
      float4 a = pu[j], b = pi[j], c = pb[j];
      v[4 * j + 0] = a.x + b.x + c.x;
      v[4 * j + 1] = a.y + b.y + c.y;
      v[4 * j + 2] = a.z + b.z + c.z;
      v[4 * j + 3] = a.w + b.w + c.w;
    }
    float s1 = 0.0f, s2 = 0.0f;
#pragma unroll
    for (int j = 0; j < 64; ++j) { s1 += v[j]; s2 += v[j] * v[j]; }
    float mu = s1 * (1.0f / 64.0f);
    float var = s2 * (1.0f / 64.0f) - mu * mu;
    float rs = rsqrtf(var + LN_EPS);
#pragma unroll
    for (int j = 0; j < 64; ++j) {
      float h = (v[j] - mu) * rs * g1[j] + be1[j];
      v[j] = h > 0.0f ? h : 0.0f;
    }
  } else {
#pragma unroll
    for (int j = 0; j < 64; ++j) v[j] = 0.0f;
  }
  {
    _Float16* rowp = &hbuf[tid * 72];
#pragma unroll
    for (int j0 = 0; j0 < 64; j0 += 8) {
      v8h h;
#pragma unroll
      for (int j = 0; j < 8; ++j) h[j] = (_Float16)v[j0 + j];
      *(v8h*)(rowp + j0) = h;  // 16B ds_store
    }
  }
  __syncthreads();

  // ---- phase 2: h2 = relu(LN(h1 @ W2 + b2)); out = h2 @ W3 + b3 ----
  v16h bf[8];  // [t*2+q]
#pragma unroll
  for (int f = 0; f < 8; ++f)
    bf[f] = ((const v16h*)W2s)[f * 32 + lane];

  float b2v[4], g2v[4], be2v[4], w3v[4];
#pragma unroll
  for (int t = 0; t < 4; ++t) {
    int n = t * 16 + (lane & 15);
    b2v[t] = b2[n];
    g2v[t] = g2[n];
    be2v[t] = be2[n];
    w3v[t] = W3[n];
  }
  float b3v = b3[0];

#pragma unroll
  for (int sub = 0; sub < 2; ++sub) {
    int m = 2 * wave + sub;                       // M-tile (this wave's rows)
    const _Float16* ap = &hbuf[(16 * m + (lane & 15)) * 72];
    v8f c[4] = {};
#pragma unroll
    for (int q = 0; q < 2; ++q) {
      // A-operand, 16-bit 16x32 layout: lanes 0-15 hold K {0-7,16-23},
      // lanes 16-31 hold K {8-15,24-31} (offset by q*32)
      int k0 = q * 32 + ((lane < 16) ? 0 : 8);
      v8h lo = *(const v8h*)(ap + k0);
      v8h hi = *(const v8h*)(ap + k0 + 16);
      v16h a = __builtin_shufflevector(lo, hi, 0, 1, 2, 3, 4, 5, 6, 7,
                                       8, 9, 10, 11, 12, 13, 14, 15);
#pragma unroll
      for (int t = 0; t < 4; ++t)
        c[t] = __builtin_amdgcn_wmma_f32_16x16x32_f16(
            false, a, false, bf[t * 2 + q], (short)0, c[t], false, false);
    }
    // epilogue in C layout: lane half-group = cols N=t*16+(lane&15),
    // VGPR r -> row M = r + (lane>=16 ? 8 : 0)
#pragma unroll
    for (int r = 0; r < 8; ++r) {
      float x0 = c[0][r] + b2v[0];
      float x1 = c[1][r] + b2v[1];
      float x2 = c[2][r] + b2v[2];
      float x3 = c[3][r] + b2v[3];
      float s1 = x0 + x1 + x2 + x3;
      float s2 = x0 * x0 + x1 * x1 + x2 * x2 + x3 * x3;
#pragma unroll
      for (int msk = 1; msk < 16; msk <<= 1) {
        s1 += __shfl_xor(s1, msk, 32);
        s2 += __shfl_xor(s2, msk, 32);
      }
      float mu = s1 * (1.0f / 64.0f);
      float var = s2 * (1.0f / 64.0f) - mu * mu;
      float rs = rsqrtf(var + LN_EPS);
      float h0 = (x0 - mu) * rs * g2v[0] + be2v[0]; h0 = h0 > 0.0f ? h0 : 0.0f;
      float h1 = (x1 - mu) * rs * g2v[1] + be2v[1]; h1 = h1 > 0.0f ? h1 : 0.0f;
      float h2 = (x2 - mu) * rs * g2v[2] + be2v[2]; h2 = h2 > 0.0f ? h2 : 0.0f;
      float h3 = (x3 - mu) * rs * g2v[3] + be2v[3]; h3 = h3 > 0.0f ? h3 : 0.0f;
      float o = h0 * w3v[0] + h1 * w3v[1] + h2 * w3v[2] + h3 * w3v[3];
#pragma unroll
      for (int msk = 1; msk < 16; msk <<= 1) o += __shfl_xor(o, msk, 32);
      if ((lane & 15) == 0) {
        int gr = blockIdx.x * 128 + 16 * m + r + ((lane >> 4) << 3);
        if (gr < B_CNT) out[gr] = o + b3v;
      }
    }
  }
}

// ---------------------------------------------------------------- dispatch
extern "C" void kernel_launch(void* const* d_in, const int* in_sizes, int n_in,
                              void* d_out, int out_size, void* d_ws, size_t ws_size,
                              hipStream_t stream) {
  const float* uE  = (const float*)d_in[0];
  const float* iE  = (const float*)d_in[1];
  const float* uH  = (const float*)d_in[2];
  const float* iH  = (const float*)d_in[3];
  const float* W1  = (const float*)d_in[4];
  const float* b1  = (const float*)d_in[5];
  const float* g1  = (const float*)d_in[6];
  const float* be1 = (const float*)d_in[7];
  const float* W2  = (const float*)d_in[8];
  const float* b2  = (const float*)d_in[9];
  const float* g2  = (const float*)d_in[10];
  const float* be2 = (const float*)d_in[11];
  const float* W3  = (const float*)d_in[12];
  const float* b3  = (const float*)d_in[13];
  const float* vals = (const float*)d_in[14];
  const int* rows   = (const int*)d_in[15];
  const int* cols   = (const int*)d_in[16];
  const int* uIdx   = (const int*)d_in[17];
  const int* sIdx   = (const int*)d_in[18];
  float* out = (float*)d_out;

  float* ws   = (float*)d_ws;
  float* embA = ws;
  float* embB = embA + (size_t)N_CNT * D_DIM;
  float* Tu   = embB + (size_t)N_CNT * D_DIM;
  float* Ti   = Tu + R_CNT * H_DIM;
  float* Mu   = Ti + R_CNT * H_DIM;
  float* Mi   = Mu + D_DIM * H_DIM;
  float* Pu   = Mi + D_DIM * H_DIM;
  float* Pi   = Pu + U_CNT * H_DIM;
  _Float16* W2s = (_Float16*)(Pi + (size_t)S_CNT * H_DIM);

  const int nd = N_CNT * D_DIM;
  const int gb = (nd + 255) / 256;
  const int eb = (E_CNT * 32 + 255) / 256;

  concat_kernel<<<gb, 256, 0, stream>>>(uE, iE, embA);

  zero_kernel<<<gb, 256, 0, stream>>>(embB, nd);
  spmm_scatter<<<eb, 256, 0, stream>>>(embA, embB, vals, rows, cols);
  zero_kernel<<<gb, 256, 0, stream>>>(embA, nd);
  spmm_scatter<<<eb, 256, 0, stream>>>(embB, embA, vals, rows, cols);
  zero_kernel<<<gb, 256, 0, stream>>>(embB, nd);
  spmm_scatter<<<eb, 256, 0, stream>>>(embA, embB, vals, rows, cols);

  tmat_kernel<<<(2 * R_CNT * H_DIM + 255) / 256, 256, 0, stream>>>(uH, iH, W1, Tu, Ti);
  mmat_kernel<<<(2 * D_DIM * H_DIM + 255) / 256, 256, 0, stream>>>(uH, iH, Tu, Ti, Mu, Mi);
  proj_kernel<<<(N_CNT * H_DIM + 255) / 256, 256, 0, stream>>>(embB, Mu, Mi, Pu, Pi);
  w2_swizzle<<<16, 256, 0, stream>>>(W2, W2s);

  mlp_kernel<<<(B_CNT + 127) / 128, 128, 0, stream>>>(
      Pu, Pi, W2s, uIdx, sIdx, b1, g1, be1, b2, g2, be2, W3, b3, out);
}
